// MessageLayer_2877628088536
// MI455X (gfx1250) — matrix operand order repaired and checked by reading, not compiled, not run on hardware
//
#include <hip/hip_runtime.h>
#include <hip/hip_bf16.h>
#include <stdint.h>

#define NN   100000
#define MM   500000
#define FF   128
#define HH   3
#define HDD  256
#define K2F  256           // 2*F
#define TILE_M 64
#define FEA_STRIDE 264     // 256 + 8 ushort pad (keeps 16B alignment, spreads banks)

typedef __bf16 bf16_t;
typedef bf16_t bf16x16 __attribute__((ext_vector_type(16)));
typedef float  f32x8   __attribute__((ext_vector_type(8)));

union BFrag { bf16x16 v; uint4 q[2]; unsigned short u[16]; };

__device__ inline unsigned short f2bf(float f) {
  unsigned int x = __float_as_uint(f);
  unsigned int r = x + 0x7FFFu + ((x >> 16) & 1u);   // round-to-nearest-even
  return (unsigned short)(r >> 16);
}
__device__ inline unsigned int fenc(float f) {       // monotone float->uint
  unsigned int b = __float_as_uint(f);
  return (b & 0x80000000u) ? ~b : (b | 0x80000000u);
}
__device__ inline float fdec(unsigned int e) {
  unsigned int b = (e & 0x80000000u) ? (e & 0x7FFFFFFFu) : ~e;
  return __uint_as_float(b);
}
__device__ inline f32x8 zero8() {
  f32x8 z;
#pragma unroll
  for (int i = 0; i < 8; ++i) z[i] = 0.f;
  return z;
}

// Gather concat(x[self], x[nbr]) rows (bf16) for a 64-edge tile into LDS.
__device__ inline void gather_fea(const unsigned short* __restrict__ elemBf,
                                  const int* __restrict__ selfIdx,
                                  const int* __restrict__ nbrIdx,
                                  int tileBase,
                                  unsigned short (*fea)[FEA_STRIDE]) {
  const int t  = threadIdx.x;       // 0..255
  const int le = t >> 2;            // local edge 0..63
  const int q  = t & 3;             // quarter of the 256-wide row
  const int e  = tileBase + le;
  const bool valid = (e < MM);
  int node = 0;
  if (valid) node = (q < 2) ? selfIdx[e] : nbrIdx[e];
  const uint4* s4 = (const uint4*)(elemBf + (size_t)node * FF + (q & 1) * 64);
  uint4* dst = (uint4*)(&fea[le][q * 64]);
#pragma unroll
  for (int i = 0; i < 8; ++i) {
    uint4 v;
    if (valid) v = s4[i]; else { v.x = v.y = v.z = v.w = 0u; }
    dst[i] = v;
  }
}

// 64x256(K) @ 256x128 per-wave slice: wave covers rows mrowBase..+16, cols nbase..+128.
// wrow: bf16 weights, layout [col][256] (pre-transposed), L2-resident.
__device__ inline void gemm_256(const unsigned short (*fea)[FEA_STRIDE],
                                const unsigned short* __restrict__ wrow,
                                f32x8 acc[8], int mrowBase, int nbase, int lane) {
  const int mrow  = mrowBase + (lane & 15);
  const int khalf = lane >> 4;
#pragma unroll
  for (int kc = 0; kc < 256; kc += 32) {
    BFrag a;   // A 16x32 bf16 fragment per ISA layout
    a.q[0] = *(const uint4*)(&fea[mrow][kc + khalf * 8]);
    a.q[1] = *(const uint4*)(&fea[mrow][kc + 16 + khalf * 8]);
#pragma unroll
    for (int nt = 0; nt < 8; ++nt) {
      const int col = nbase + nt * 16 + (lane & 15);
      BFrag b;   // B 32x16: lane = column, K = 8*khalf+0..7 then 16+8*khalf+0..7
      const uint4* bp = (const uint4*)(wrow + (size_t)col * 256 + kc + khalf * 8);
      b.q[0] = bp[0]; b.q[1] = bp[2];
      acc[nt] = __builtin_amdgcn_wmma_f32_16x16x32_bf16(
          false, a.v, false, b.v, (short)0, acc[nt], false, false);
    }
  }
}

// ---------------- Kernel 0: weight transpose + bf16 conversion ----------------
__global__ void prep_kernel(const float* __restrict__ gateW1, const float* __restrict__ msgW1,
                            const float* __restrict__ msgWo,  const float* __restrict__ elemIn,
                            unsigned short* __restrict__ w1t, unsigned short* __restrict__ msgWoT,
                            unsigned short* __restrict__ elemBf) {
  const size_t stride = (size_t)gridDim.x * blockDim.x;
  const size_t tid0 = blockIdx.x * (size_t)blockDim.x + threadIdx.x;
  // w1t[(net*H+h)*HD+col][k]  <-  W1[h][k][col]
  const size_t w1total = (size_t)2 * HH * HDD * K2F;
  for (size_t i = tid0; i < w1total; i += stride) {
    const int k = (int)(i & 255);
    const size_t tmp = i >> 8;
    const int col = (int)(tmp % HDD);
    const int nhh = (int)(tmp / HDD);
    const int net = nhh / HH, h = nhh % HH;
    const float* src = net ? msgW1 : gateW1;
    w1t[i] = f2bf(src[((size_t)h * K2F + k) * HDD + col]);
  }
  // msgWoT[h][col(F)][k(HD)]  <-  msg_Wo[h][k][col]
  const size_t wototal = (size_t)HH * FF * HDD;
  for (size_t i = tid0; i < wototal; i += stride) {
    const int k = (int)(i % HDD);
    const size_t tmp = i / HDD;
    const int col = (int)(tmp % FF);
    const int h = (int)(tmp / FF);
    msgWoT[i] = f2bf(msgWo[((size_t)h * HDD + k) * FF + col]);
  }
  const size_t et = (size_t)NN * FF;
  for (size_t i = tid0; i < et; i += stride) elemBf[i] = f2bf(elemIn[i]);
}

// ---------------- Kernel 1: init (out = residual; zero accumulators) ----------------
__global__ void init_kernel(float* __restrict__ out, const float* __restrict__ elemIn,
                            unsigned int* __restrict__ gmaxEnc, float* __restrict__ den,
                            float* __restrict__ sums, float* __restrict__ sumsq) {
  const size_t nf = (size_t)NN * FF;
  const size_t nh = (size_t)NN * HH;
  const size_t st = (size_t)2 * HH * HDD;
  const size_t stride = (size_t)gridDim.x * blockDim.x;
  for (size_t i = blockIdx.x * (size_t)blockDim.x + threadIdx.x; i < nf; i += stride) {
    out[i] = elemIn[i];
    if (i < nh) { gmaxEnc[i] = fenc(-3.0e38f); den[i] = 0.f; }
    if (i < st) { sums[i] = 0.f; sumsq[i] = 0.f; }
  }
}

// ---------------- Kernel 2: BN batch statistics (both nets, all heads) ----------------
__global__ __launch_bounds__(256)
void stats_kernel(const unsigned short* __restrict__ elemBf, const unsigned short* __restrict__ w1t,
                  const int* __restrict__ selfIdx, const int* __restrict__ nbrIdx,
                  float* __restrict__ sums, float* __restrict__ sumsq) {
  __shared__ unsigned short fea[TILE_M][FEA_STRIDE];
  __shared__ float colsum[256], colsq[256];
  const int t = threadIdx.x, wave = t >> 5, lane = t & 31;
  const int tileBase = blockIdx.x * TILE_M;
  const int nh = blockIdx.y;                          // net*H + head, 0..5
  const unsigned short* w = w1t + (size_t)nh * HDD * 256;
  if (t < 64) __builtin_prefetch(w + (size_t)t * 1024, 0, 0);
  gather_fea(elemBf, selfIdx, nbrIdx, tileBase, fea);
  if (t < 256) { colsum[t] = 0.f; colsq[t] = 0.f; }
  __syncthreads();
  f32x8 acc[8];
#pragma unroll
  for (int i = 0; i < 8; ++i) acc[i] = zero8();
  const int mrowBase = (wave & 3) * 16;
  const int nbase = (wave >> 2) * 128;
  gemm_256(fea, w, acc, mrowBase, nbase, lane);
#pragma unroll
  for (int nt = 0; nt < 8; ++nt) {
    float s = 0.f, sq = 0.f;
#pragma unroll
    for (int r = 0; r < 8; ++r) { const float v = acc[nt][r]; s += v; sq += v * v; }
    s  += __shfl_xor(s, 16);
    sq += __shfl_xor(sq, 16);
    if (lane < 16) {
      const int col = nbase + nt * 16 + lane;
      atomicAdd(&colsum[col], s);
      atomicAdd(&colsq[col], sq);
    }
  }
  __syncthreads();
  if (t < 256) {
    atomicAdd(&sums [nh * HDD + t], colsum[t]);
    atomicAdd(&sumsq[nh * HDD + t], colsq[t]);
  }
}

// ---------------- Kernel 3: fold BN into per-column scale/shift ----------------
__global__ void bn_finalize(const float* __restrict__ sums, const float* __restrict__ sumsq,
                            const float* __restrict__ gateG, const float* __restrict__ gateBeta,
                            const float* __restrict__ msgG,  const float* __restrict__ msgBeta,
                            float* __restrict__ scale, float* __restrict__ shift) {
  const int i = blockIdx.x * blockDim.x + threadIdx.x;
  if (i >= 2 * HH * HDD) return;
  const int net = i / (HH * HDD);
  const int rem = i % (HH * HDD);
  const float* g  = net ? msgG    : gateG;
  const float* be = net ? msgBeta : gateBeta;
  const float inv = 1.f / (float)MM;
  const float mu  = sums[i] * inv;
  const float var = sumsq[i] * inv - mu * mu;
  const float sc  = g[rem] * rsqrtf(var + 1e-5f);
  scale[i] = sc;
  shift[i] = be[rem] - mu * sc;
}

// ---------------- Kernel 4: gate logits + atomic segment max ----------------
__global__ __launch_bounds__(256)
void gate_kernel(const unsigned short* __restrict__ elemBf, const unsigned short* __restrict__ w1t,
                 const int* __restrict__ selfIdx, const int* __restrict__ nbrIdx,
                 const float* __restrict__ scale, const float* __restrict__ shift,
                 const float* __restrict__ gateWo, const float* __restrict__ gateBo,
                 float* __restrict__ gateLogit, unsigned int* __restrict__ gmaxEnc) {
  __shared__ unsigned short fea[TILE_M][FEA_STRIDE];
  __shared__ float edgeAcc[TILE_M];
  const int t = threadIdx.x, wave = t >> 5, lane = t & 31;
  const int tileBase = blockIdx.x * TILE_M;
  const int head = blockIdx.y;
  const int nh = head;                                  // gate net
  const unsigned short* w = w1t + (size_t)nh * HDD * 256;
  gather_fea(elemBf, selfIdx, nbrIdx, tileBase, fea);
  if (t < TILE_M) edgeAcc[t] = 0.f;
  __syncthreads();
  f32x8 acc[8];
#pragma unroll
  for (int i = 0; i < 8; ++i) acc[i] = zero8();
  const int mrowBase = (wave & 3) * 16;
  const int nbase = (wave >> 2) * 128;
  gemm_256(fea, w, acc, mrowBase, nbase, lane);
  const float* sc = scale + nh * HDD;
  const float* sh = shift + nh * HDD;
  const float* wo = gateWo + head * HDD;
  float rowpart[8];
#pragma unroll
  for (int r = 0; r < 8; ++r) rowpart[r] = 0.f;
#pragma unroll
  for (int nt = 0; nt < 8; ++nt) {
    const int col = nbase + nt * 16 + (lane & 15);
    const float s1 = sc[col], s2 = sh[col], wv = wo[col];
#pragma unroll
    for (int r = 0; r < 8; ++r) {
      const float v = acc[nt][r] * s1 + s2;
      const float sv = v / (1.f + expf(-v));            // SiLU
      rowpart[r] += sv * wv;
    }
  }
#pragma unroll
  for (int r = 0; r < 8; ++r) {
    float x = rowpart[r];
    x += __shfl_xor(x, 1); x += __shfl_xor(x, 2);
    x += __shfl_xor(x, 4); x += __shfl_xor(x, 8);
    if ((lane & 15) == 0)
      atomicAdd(&edgeAcc[mrowBase + r + 8 * (lane >> 4)], x);
  }
  __syncthreads();
  if (t < TILE_M) {
    const int e = tileBase + t;
    if (e < MM) {
      const float logit = edgeAcc[t] + gateBo[head];
      gateLogit[(size_t)e * HH + head] = logit;
      atomicMax(&gmaxEnc[(size_t)selfIdx[e] * HH + head], fenc(logit));
    }
  }
}

// ---------------- Kernel 5: softmax numerator + atomic denominator ----------------
__global__ void softmax_kernel(const float* __restrict__ gateLogit, const unsigned int* __restrict__ gmaxEnc,
                               const int* __restrict__ selfIdx, const int* __restrict__ nbrIdx,
                               const float* __restrict__ elemW, const float* __restrict__ powP,
                               float* __restrict__ numer, float* __restrict__ den) {
  const size_t total = (size_t)MM * HH;
  const size_t stride = (size_t)gridDim.x * blockDim.x;
  for (size_t i = blockIdx.x * (size_t)blockDim.x + threadIdx.x; i < total; i += stride) {
    const int h = (int)(i % HH);
    const int e = (int)(i / HH);
    const int sn = selfIdx[e], nn = nbrIdx[e];
    const float m = fdec(gmaxEnc[(size_t)sn * HH + h]);
    const float nu = powf(elemW[nn], powP[h]) * expf(gateLogit[i] - m);
    numer[i] = nu;
    atomicAdd(&den[(size_t)sn * HH + h], nu);
  }
}

// ---------------- Kernel 6: message net + gated segment-sum scatter ----------------
__global__ __launch_bounds__(256)
void msg_kernel(const unsigned short* __restrict__ elemBf, const unsigned short* __restrict__ w1t,
                const unsigned short* __restrict__ msgWoT,
                const int* __restrict__ selfIdx, const int* __restrict__ nbrIdx,
                const float* __restrict__ scale, const float* __restrict__ shift,
                const float* __restrict__ msgBo,
                const float* __restrict__ numer, const float* __restrict__ den,
                float* __restrict__ out) {
  __shared__ unsigned short fea[TILE_M][FEA_STRIDE];
  __shared__ float gfac[TILE_M];
  __shared__ int   nodeIds[TILE_M];
  const int t = threadIdx.x, wave = t >> 5, lane = t & 31;
  const int tileBase = blockIdx.x * TILE_M;
  const int head = blockIdx.y;
  const int nh = HH + head;                             // msg net
  const unsigned short* w = w1t + (size_t)nh * HDD * 256;
  gather_fea(elemBf, selfIdx, nbrIdx, tileBase, fea);
  if (t < TILE_M) {
    const int e = tileBase + t;
    if (e < MM) {
      const int sn = selfIdx[e];
      nodeIds[t] = sn;
      gfac[t] = numer[(size_t)e * HH + head] /
                (den[(size_t)sn * HH + head] + 1e-10f) * (1.f / (float)HH);
    } else { nodeIds[t] = 0; gfac[t] = 0.f; }
  }
  __syncthreads();
  f32x8 acc[8];
#pragma unroll
  for (int i = 0; i < 8; ++i) acc[i] = zero8();
  const int mrowBase = (wave & 3) * 16;
  const int nbase = (wave >> 2) * 128;
  gemm_256(fea, w, acc, mrowBase, nbase, lane);
  const float* sc = scale + nh * HDD;
  const float* sh = shift + nh * HDD;
#pragma unroll
  for (int nt = 0; nt < 8; ++nt) {
    const int col = nbase + nt * 16 + (lane & 15);
    const float s1 = sc[col], s2 = sh[col];
#pragma unroll
    for (int r = 0; r < 8; ++r) {
      const float v = acc[nt][r] * s1 + s2;
      acc[nt][r] = v / (1.f + expf(-v));                // SiLU
    }
  }
  __syncthreads();    // all fea reads for GEMM1 complete before overwrite
#pragma unroll
  for (int nt = 0; nt < 8; ++nt) {
    const int col = nbase + nt * 16 + (lane & 15);
#pragma unroll
    for (int r = 0; r < 8; ++r) {
      const int row = mrowBase + r + 8 * (lane >> 4);
      fea[row][col] = f2bf(acc[nt][r]);
    }
  }
  __syncthreads();
  // GEMM2: (64x256) @ (256x128) with msgWoT[head] laid out [F][HD]
  const unsigned short* w2 = msgWoT + (size_t)head * FF * HDD;
  f32x8 acc2[4];
#pragma unroll
  for (int i = 0; i < 4; ++i) acc2[i] = zero8();
  const int n2base = (wave >> 2) * 64;
  const int mrow = mrowBase + (lane & 15);
  const int khalf = lane >> 4;
#pragma unroll
  for (int kc = 0; kc < 256; kc += 32) {
    BFrag a;
    a.q[0] = *(const uint4*)(&fea[mrow][kc + khalf * 8]);
    a.q[1] = *(const uint4*)(&fea[mrow][kc + 16 + khalf * 8]);
#pragma unroll
    for (int nt = 0; nt < 4; ++nt) {
      const int col = n2base + nt * 16 + (lane & 15);
      BFrag b;
      const uint4* bp = (const uint4*)(w2 + (size_t)col * HDD + kc + khalf * 8);
      b.q[0] = bp[0]; b.q[1] = bp[2];
      acc2[nt] = __builtin_amdgcn_wmma_f32_16x16x32_bf16(
          false, a.v, false, b.v, (short)0, acc2[nt], false, false);
    }
  }
  const float* bo = msgBo + head * FF;
#pragma unroll
  for (int nt = 0; nt < 4; ++nt) {
    const int col = n2base + nt * 16 + (lane & 15);
    const float b0 = bo[col];
#pragma unroll
    for (int r = 0; r < 8; ++r) {
      const int row = mrowBase + r + 8 * (lane >> 4);
      const int e = tileBase + row;
      if (e < MM) {
        const float val = (acc2[nt][r] + b0) * gfac[row];
        atomicAdd(&out[(size_t)nodeIds[row] * FF + col], val);
      }
    }
  }
}

extern "C" void kernel_launch(void* const* d_in, const int* in_sizes, int n_in,
                              void* d_out, int out_size, void* d_ws, size_t ws_size,
                              hipStream_t stream) {
  (void)in_sizes; (void)n_in; (void)out_size; (void)ws_size;
  const float* elemW    = (const float*)d_in[0];
  const float* elemIn   = (const float*)d_in[1];
  const float* gateW1   = (const float*)d_in[2];
  const float* gateG    = (const float*)d_in[4];
  const float* gateBeta = (const float*)d_in[5];
  const float* gateWo   = (const float*)d_in[6];
  const float* gateBo   = (const float*)d_in[7];
  const float* msgW1    = (const float*)d_in[8];
  const float* msgG     = (const float*)d_in[10];
  const float* msgBeta  = (const float*)d_in[11];
  const float* msgWo    = (const float*)d_in[12];
  const float* msgBo    = (const float*)d_in[13];
  const float* powP     = (const float*)d_in[14];
  const int*   selfIdx  = (const int*)d_in[15];
  const int*   nbrIdx   = (const int*)d_in[16];
  float* out = (float*)d_out;

  char* ws = (char*)d_ws;
  size_t off = 0;
  auto take = [&](size_t bytes) -> char* {
    char* p = ws + off;
    off = (off + bytes + 255) & ~(size_t)255;
    return p;
  };
  unsigned short* elemBf = (unsigned short*)take((size_t)NN * FF * 2);
  unsigned short* w1t    = (unsigned short*)take((size_t)2 * HH * HDD * K2F * 2);
  unsigned short* msgWoT = (unsigned short*)take((size_t)HH * FF * HDD * 2);
  float* sums      = (float*)take((size_t)2 * HH * HDD * 4);
  float* sumsq     = (float*)take((size_t)2 * HH * HDD * 4);
  float* scaleB    = (float*)take((size_t)2 * HH * HDD * 4);
  float* shiftB    = (float*)take((size_t)2 * HH * HDD * 4);
  float* gateLogit = (float*)take((size_t)MM * HH * 4);
  float* numer     = (float*)take((size_t)MM * HH * 4);
  unsigned int* gmaxEnc = (unsigned int*)take((size_t)NN * HH * 4);
  float* den       = (float*)take((size_t)NN * HH * 4);

  const int tiles = (MM + TILE_M - 1) / TILE_M;

  prep_kernel<<<2048, 256, 0, stream>>>(gateW1, msgW1, msgWo, elemIn, w1t, msgWoT, elemBf);
  init_kernel<<<2048, 256, 0, stream>>>(out, elemIn, gmaxEnc, den, sums, sumsq);
  stats_kernel<<<dim3(tiles, 2 * HH), 256, 0, stream>>>(elemBf, w1t, selfIdx, nbrIdx, sums, sumsq);
  bn_finalize<<<(2 * HH * HDD + 255) / 256, 256, 0, stream>>>(sums, sumsq, gateG, gateBeta,
                                                              msgG, msgBeta, scaleB, shiftB);
  gate_kernel<<<dim3(tiles, HH), 256, 0, stream>>>(elemBf, w1t, selfIdx, nbrIdx,
                                                   scaleB, shiftB, gateWo, gateBo,
                                                   gateLogit, gmaxEnc);
  softmax_kernel<<<2048, 256, 0, stream>>>(gateLogit, gmaxEnc, selfIdx, nbrIdx,
                                           elemW, powP, numer, den);
  msg_kernel<<<dim3(tiles, HH), 256, 0, stream>>>(elemBf, w1t, msgWoT, selfIdx, nbrIdx,
                                                  scaleB, shiftB, msgBo, numer, den, out);
}
